// MultiHeadAttentionBlock_83279415870173
// MI455X (gfx1250) — matrix operand order repaired and checked by reading, not compiled, not run on hardware
//
#include <hip/hip_runtime.h>

typedef __attribute__((ext_vector_type(16))) _Float16     v16h;
typedef __attribute__((ext_vector_type(8)))  float        v8f;
typedef __attribute__((ext_vector_type(4)))  unsigned int u32x4;
typedef __attribute__((ext_vector_type(8)))  int          i32x8;
typedef __attribute__((ext_vector_type(4)))  int          i32x4;

#define HIDDEN  1024
#define SEQ     2048
#define BATCH   4
#define HEADS   16
#define HDIM    64
#define ROWS    (BATCH * SEQ)   // 8192
#define KCHUNK  64              // K rows staged per TDM transfer

__device__ __forceinline__ v8f wmma_f16(v16h a, v16h b, v8f c) {
    // 8 args: (neg_a, A, neg_b, B, c_mod, C, reuse_a, reuse_b)
    return __builtin_amdgcn_wmma_f32_16x16x32_f16(
        false, a, false, b, (short)0, c, false, false);
}

__device__ __forceinline__ unsigned rfl(unsigned v) {
    return (unsigned)__builtin_amdgcn_readfirstlane((int)v);
}

// Issue a TDM 2D tile load (Global -> LDS), D# built per CDNA5 ISA
//  - tile: tile_d1 rows x tile_d0 elements (f16, data_size code 1 = 2 bytes)
//  - tensor row stride: stride0 elements; LDS dest: packed row-major
__device__ __forceinline__ void tdm_load_2d(unsigned lds_off, unsigned long long gaddr,
                                            unsigned tile_d0, unsigned tile_d1,
                                            unsigned stride0) {
    u32x4 g0;
    g0.x = rfl(1u);                                            // count=1, user mode
    g0.y = rfl(lds_off);                                       // lds_addr [63:32]
    g0.z = rfl((unsigned)(gaddr & 0xFFFFFFFFull));             // global_addr lo
    g0.w = rfl(((unsigned)((gaddr >> 32) & 0x1FFFFFFull))      // global_addr [56:32]
               | (2u << 30));                                  // type=2 ("image")
    i32x8 g1;
    g1[0] = (int)rfl(1u << 16);                    // wg_mask=0 | data_size=1 (2B)
    g1[1] = (int)rfl((1024u & 0xFFFFu) << 16);     // atomic_addr=0 | tensor_dim0 lo16
    g1[2] = (int)rfl(((1024u >> 16) & 0xFFFFu) | ((1024u & 0xFFFFu) << 16)); // d0 hi | d1 lo
    g1[3] = (int)rfl(((1024u >> 16) & 0xFFFFu) | (tile_d0 << 16));           // d1 hi | tile_dim0
    g1[4] = (int)rfl(tile_d1 & 0xFFFFu);           // tile_dim1 | tile_dim2=0
    g1[5] = (int)rfl(stride0);                     // tensor_dim0_stride lo32
    g1[6] = 0;                                     // stride0 hi16 | stride1 lo16
    g1[7] = 0;                                     // stride1 hi32 (unused, 2D)
    i32x4 z4 = {0, 0, 0, 0};                       // groups 2/3: unused (2D tensor)
    i32x8 z8 = {0, 0, 0, 0, 0, 0, 0, 0};
    // amdgpu-toolchain (clang-23) 6-arg form
    __builtin_amdgcn_tensor_load_to_lds(g0, g1, z4, z4, z8, 0);
}

// Load a 16x32 f16 A-matrix tile in the ISA-mandated layout:
// lane L: m = L&15 ; lanes 0-15 hold K in {0..7,16..23}, lanes 16-31 K in {8..15,24..31}
__device__ __forceinline__ v16h load_a16(const _Float16* __restrict__ base, int ld) {
    const int L  = threadIdx.x & 31;
    const int m  = L & 15;
    const int kb = (L >> 4) << 3;           // 0 or 8
    const _Float16* p = base + (size_t)m * ld;
    v16h a;
#pragma unroll
    for (int v = 0; v < 8; ++v) {
        int k0 = (v < 4) ? (kb + 2 * v) : (16 + kb + 2 * (v - 4));
        a[2 * v]     = p[k0];
        a[2 * v + 1] = p[k0 + 1];
    }
    return a;
}

// ---------------------------------------------------------------- converters
__global__ void cvt_f32_f16(const float* __restrict__ in, _Float16* __restrict__ out, int n) {
    int i = blockIdx.x * 256 + threadIdx.x;
    if (i < n) out[i] = (_Float16)in[i];
}

// Wt[k][n] = W[n][k]  (f32 -> f16), W is [out=1024, in=1024]
__global__ void cvt_transpose(const float* __restrict__ W, _Float16* __restrict__ Wt) {
    int i = blockIdx.x * 256 + threadIdx.x;     // 0 .. 1024*1024-1
    int k = i >> 10, n = i & 1023;
    Wt[i] = (_Float16)W[(size_t)n * HIDDEN + k];
}

// ---------------------------------------------------------------- WMMA GEMM
// C[M=8192, N=1024] = A[8192,1024](f16) * Bt[1024,1024](f16, k-major) + bias
// Workgroup = 4 waves; each wave owns a 16x64 tile (64x64 per block).
// B panel (KCHUNK x 64) staged into LDS by the Tensor Data Mover, double-buffered,
// shared by all 4 waves. A tiles streamed from global (b128 per lane).
// mode 0: f16 [B, heads, S, 64] (Q, V) | mode 1: f16 [B, heads, 64, S] (Kt)
// mode 2: f32 flat [8192, 1024] (output projection)
__global__ __launch_bounds__(128)
void gemm_wmma(const _Float16* __restrict__ A, const _Float16* __restrict__ Bt,
               const float* __restrict__ bias,
               _Float16* __restrict__ outh, float* __restrict__ outf, int mode) {
    __shared__ _Float16 bbuf[2][KCHUNK * 64];

    const int tid   = threadIdx.x;
    const int L     = tid & 31;
    const int wid   = tid >> 5;            // wave 0..3
    const int bid   = blockIdx.x;
    const int tn    = bid & 15;            // 16 n-tiles of 64
    const int tmb   = bid >> 4;            // 128 m-blocks of 64 rows
    const int mbase = tmb * 64 + wid * 16;
    const int nbase = tn * 64;

    // prime the pipeline: TDM chunk 0 -> buffer 0 (wave 0 only; EXEC-independent op)
    if (wid == 0) {
        tdm_load_2d((unsigned)(uintptr_t)&bbuf[0][0],
                    (unsigned long long)(uintptr_t)(Bt + nbase),
                    64, KCHUNK, HIDDEN);
    }

    v8f acc[4] = {};
    const _Float16* arow = A + (size_t)mbase * HIDDEN;

    for (int kc = 0; kc < HIDDEN; kc += KCHUNK) {
        const int cur = (kc / KCHUNK) & 1;
        if (wid == 0) {
            if (kc + KCHUNK < HIDDEN) {
                tdm_load_2d((unsigned)(uintptr_t)&bbuf[cur ^ 1][0],
                            (unsigned long long)(uintptr_t)
                                (Bt + (size_t)(kc + KCHUNK) * HIDDEN + nbase),
                            64, KCHUNK, HIDDEN);
                __builtin_amdgcn_s_wait_tensorcnt((short)1);  // oldest (current) done
            } else {
                __builtin_amdgcn_s_wait_tensorcnt((short)0);
            }
        }
        __syncthreads();                    // buffer `cur` visible to all waves

        if (kc + KCHUNK < HIDDEN)
            __builtin_prefetch(arow + kc + KCHUNK, 0, 1);     // global_prefetch_b8

#pragma unroll
        for (int ks = 0; ks < KCHUNK; ks += 32) {
            v16h a = load_a16(arow + kc + ks, HIDDEN);
            const int hb = (L >> 4) << 3;
            const _Float16* bcol = &bbuf[cur][ks * 64 + (L & 15)]; // lane L = n col
#pragma unroll
            for (int t = 0; t < 4; ++t) {
                v16h b;
#pragma unroll
                for (int s = 0; s < 16; ++s) {
                    int kk = (s < 8) ? (hb + s) : (8 + hb + s);
                    b[s] = bcol[kk * 64 + t * 16];
                }
                acc[t] = wmma_f16(a, b, acc[t]);
            }
        }
        __syncthreads();                    // done reading before TDM reuses buffer
    }

#pragma unroll
    for (int t = 0; t < 4; ++t) {
#pragma unroll
        for (int j = 0; j < 8; ++j) {
            int m  = j + ((L >> 4) << 3);  // C layout: row = VGPR j (+8 for hi half)
            int n  = L & 15;
            int gm = mbase + m;
            int gn = nbase + t * 16 + n;
            float val = acc[t][j] + bias[gn];
            if (mode == 2) {
                outf[(size_t)gm * HIDDEN + gn] = val;
            } else {
                int bb = gm >> 11, s = gm & (SEQ - 1);
                int hd = gn >> 6,  d = gn & (HDIM - 1);
                if (mode == 0)
                    outh[(((size_t)(bb * HEADS + hd) * SEQ) + s) * HDIM + d] = (_Float16)val;
                else
                    outh[(((size_t)(bb * HEADS + hd) * HDIM) + d) * SEQ + s] = (_Float16)val;
            }
        }
    }
}

// ---------------------------------------------------------------- attention
// One wave per (b, head, 16-query tile). Flash-attention over S in 32-key chunks.
// Q: [bh, S, 64] f16 ; Kt: [bh, 64, S] f16 ; V: [bh, S, 64] f16 ; out: f16 [B,S,1024]
__global__ __launch_bounds__(32)
void attn_wmma(const _Float16* __restrict__ Qh, const _Float16* __restrict__ KtH,
               const _Float16* __restrict__ Vh, _Float16* __restrict__ attnh) {
    __shared__ _Float16 pl[16 * 34];       // P tile re-layout buffer (padded stride)
    const int L   = threadIdx.x;
    const int bid = blockIdx.x;
    const int qt  = bid & 127;             // 128 query tiles
    const int bh  = bid >> 7;              // 64 (batch, head) pairs

    const _Float16* q  = Qh  + (size_t)bh * SEQ * HDIM + (size_t)qt * 16 * HDIM;
    const _Float16* kt = KtH + (size_t)bh * HDIM * SEQ;
    const _Float16* v  = Vh  + (size_t)bh * SEQ * HDIM;

    v16h qa0 = load_a16(q,      HDIM);     // d = 0..31
    v16h qa1 = load_a16(q + 32, HDIM);     // d = 32..63

    v8f o[4] = {};
    float rm[8], rl[8];
#pragma unroll
    for (int j = 0; j < 8; ++j) { rm[j] = -1e30f; rl[j] = 0.f; }

    for (int kb = 0; kb < SEQ; kb += 32) {
        // scores: two 16x16 tiles, each accumulated over d=0..63 (2 WMMAs each)
        const _Float16* k0p = kt + kb + (L & 15);                  // key = kb + L%16
        const int hb = (L >> 4) << 3;
        v8f s0 = {}, s1 = {};
        {
            v16h blo, bhi, blo2, bhi2;
#pragma unroll
            for (int s = 0; s < 16; ++s) {
                int kk = (s < 8) ? (hb + s) : (8 + hb + s);
                blo[s]  = k0p[(size_t)kk * SEQ];
                bhi[s]  = k0p[(size_t)(32 + kk) * SEQ];
                blo2[s] = k0p[(size_t)kk * SEQ + 16];
                bhi2[s] = k0p[(size_t)(32 + kk) * SEQ + 16];
            }
            s0 = wmma_f16(qa0, blo, s0);
            s0 = wmma_f16(qa1, bhi, s0);
            s1 = wmma_f16(qa0, blo2, s1);
            s1 = wmma_f16(qa1, bhi2, s1);
        }

        // online softmax; row m = VGPR j (+8 for lanes 16-31), cols striped on 16 lanes
#pragma unroll
        for (int j = 0; j < 8; ++j) {
            float a0 = s0[j] * 0.125f;     // 1/sqrt(64)
            float a1 = s1[j] * 0.125f;
            float t = fmaxf(a0, a1);
            t = fmaxf(t, __shfl_xor(t, 1, 32));
            t = fmaxf(t, __shfl_xor(t, 2, 32));
            t = fmaxf(t, __shfl_xor(t, 4, 32));
            t = fmaxf(t, __shfl_xor(t, 8, 32));
            float mnew = fmaxf(rm[j], t);
            float f    = __expf(rm[j] - mnew);
            float p0   = __expf(a0 - mnew);
            float p1   = __expf(a1 - mnew);
            float ps   = p0 + p1;
            ps += __shfl_xor(ps, 1, 32);
            ps += __shfl_xor(ps, 2, 32);
            ps += __shfl_xor(ps, 4, 32);
            ps += __shfl_xor(ps, 8, 32);
            rl[j] = rl[j] * f + ps;
            rm[j] = mnew;
            o[0][j] *= f; o[1][j] *= f; o[2][j] *= f; o[3][j] *= f;
            int m = j + ((L >> 4) << 3);
            int n = L & 15;
            pl[m * 34 + n]      = (_Float16)p0;   // keys kb..kb+15
            pl[m * 34 + 16 + n] = (_Float16)p1;   // keys kb+16..kb+31
        }

        // P (16x32) back in A layout via LDS; then P*V for all 4 d-tiles
        v16h pa = load_a16(pl, 34);
        const _Float16* vcol = v + (size_t)kb * HDIM + (L & 15);  // lane L = d col
#pragma unroll
        for (int t4 = 0; t4 < 4; ++t4) {
            v16h vb;
#pragma unroll
            for (int s = 0; s < 16; ++s) {
                int kk = (s < 8) ? (hb + s) : (8 + hb + s);
                vb[s] = vcol[(size_t)kk * HDIM + t4 * 16];
            }
            o[t4] = wmma_f16(pa, vb, o[t4]);
        }
    }

    // epilogue: normalize and scatter into [B, S, 1024] f16 layout
    const size_t b = bh >> 4, h = bh & 15;
#pragma unroll
    for (int t4 = 0; t4 < 4; ++t4) {
#pragma unroll
        for (int j = 0; j < 8; ++j) {
            int m = j + ((L >> 4) << 3);
            int n = L & 15;
            float val = o[t4][j] / rl[j];
            size_t s = (size_t)qt * 16 + m;
            attnh[(b * SEQ + s) * HIDDEN + h * HDIM + t4 * 16 + n] = (_Float16)val;
        }
    }
}

// ---------------------------------------------------------------- residual+LN
__global__ __launch_bounds__(256)
void ln_kernel(const float* __restrict__ x, const float* __restrict__ proj,
               const float* __restrict__ g, const float* __restrict__ be,
               float* __restrict__ out) {
    const int row = blockIdx.x;
    const int tid = threadIdx.x;
    const float* xr = x    + (size_t)row * HIDDEN;
    const float* pr = proj + (size_t)row * HIDDEN;
    float vals[4];
    float s = 0.f, s2 = 0.f;
#pragma unroll
    for (int i = 0; i < 4; ++i) {
        float v = xr[tid + 256 * i] + pr[tid + 256 * i];
        vals[i] = v; s += v; s2 += v * v;
    }
#pragma unroll
    for (int mask = 1; mask < 32; mask <<= 1) {
        s  += __shfl_xor(s,  mask, 32);
        s2 += __shfl_xor(s2, mask, 32);
    }
    __shared__ float sm[8], sm2[8];
    __shared__ float mu_s, rs_s;
    int wid = tid >> 5;
    if ((tid & 31) == 0) { sm[wid] = s; sm2[wid] = s2; }
    __syncthreads();
    if (tid == 0) {
        float ts = 0.f, ts2 = 0.f;
        for (int w = 0; w < 8; ++w) { ts += sm[w]; ts2 += sm2[w]; }
        float mu  = ts * (1.f / HIDDEN);
        float var = ts2 * (1.f / HIDDEN) - mu * mu;
        mu_s = mu;
        rs_s = rsqrtf(var + 1e-5f);
    }
    __syncthreads();
    float mu = mu_s, rs = rs_s;
    float* outr = out + (size_t)row * HIDDEN;
#pragma unroll
    for (int i = 0; i < 4; ++i) {
        int c = tid + 256 * i;
        outr[c] = (vals[i] - mu) * rs * g[c] + be[c];
    }
}

// ---------------------------------------------------------------- launch
extern "C" void kernel_launch(void* const* d_in, const int* in_sizes, int n_in,
                              void* d_out, int out_size, void* d_ws, size_t ws_size,
                              hipStream_t stream) {
    (void)in_sizes; (void)n_in; (void)out_size; (void)ws_size;
    const float* x     = (const float*)d_in[0];
    const float* Wq    = (const float*)d_in[1];
    const float* bq    = (const float*)d_in[2];
    const float* Wk    = (const float*)d_in[3];
    const float* bk    = (const float*)d_in[4];
    const float* Wv    = (const float*)d_in[5];
    const float* bv    = (const float*)d_in[6];
    const float* Wo    = (const float*)d_in[7];
    const float* bo    = (const float*)d_in[8];
    const float* gamma = (const float*)d_in[9];
    const float* beta  = (const float*)d_in[10];

    char* ws = (char*)d_ws;
    _Float16* xh    = (_Float16*)(ws);                        // 16 MiB
    _Float16* WqT   = (_Float16*)(ws + ((size_t)16 << 20));   //  2 MiB
    _Float16* WkT   = (_Float16*)(ws + ((size_t)18 << 20));
    _Float16* WvT   = (_Float16*)(ws + ((size_t)20 << 20));
    _Float16* WoT   = (_Float16*)(ws + ((size_t)22 << 20));
    _Float16* Qh    = (_Float16*)(ws + ((size_t)24 << 20));   // 16 MiB
    _Float16* KtH   = (_Float16*)(ws + ((size_t)40 << 20));   // 16 MiB
    _Float16* Vh    = (_Float16*)(ws + ((size_t)56 << 20));   // 16 MiB
    _Float16* attnh = (_Float16*)(ws + ((size_t)72 << 20));   // 16 MiB
    float*    proj  = (float*)   (ws + ((size_t)24 << 20));   // 32 MiB, reuses Q/Kt
                                                              // (dead after attention)

    const int NELEM = ROWS * HIDDEN;
    cvt_f32_f16  <<<(NELEM + 255) / 256, 256, 0, stream>>>(x, xh, NELEM);
    cvt_transpose<<<(HIDDEN * HIDDEN) / 256, 256, 0, stream>>>(Wq, WqT);
    cvt_transpose<<<(HIDDEN * HIDDEN) / 256, 256, 0, stream>>>(Wk, WkT);
    cvt_transpose<<<(HIDDEN * HIDDEN) / 256, 256, 0, stream>>>(Wv, WvT);
    cvt_transpose<<<(HIDDEN * HIDDEN) / 256, 256, 0, stream>>>(Wo, WoT);

    const int GEMM_GRID = (ROWS / 64) * (HIDDEN / 64);        // 2048 blocks x 4 waves
    gemm_wmma<<<GEMM_GRID, 128, 0, stream>>>(xh, WqT, bq, Qh,  nullptr, 0);
    gemm_wmma<<<GEMM_GRID, 128, 0, stream>>>(xh, WkT, bk, KtH, nullptr, 1);
    gemm_wmma<<<GEMM_GRID, 128, 0, stream>>>(xh, WvT, bv, Vh,  nullptr, 0);

    attn_wmma<<<BATCH * HEADS * (SEQ / 16), 32, 0, stream>>>(Qh, KtH, Vh, attnh);

    gemm_wmma<<<GEMM_GRID, 128, 0, stream>>>(attnh, WoT, bo, nullptr, proj, 2);

    ln_kernel<<<ROWS, 256, 0, stream>>>(x, proj, gamma, beta, (float*)d_out);
}
